// Encoder_18691697672579
// MI455X (gfx1250) — 8-bit matrix operand order repaired and checked by reading, compiled, not run on hardware
//
#include <hip/hip_runtime.h>
#include <hip/hip_bf16.h>


#define B_   64
#define N_   16
#define I_   128
#define H_   256
#define V_   128
#define G3H  768
#define ROWS (B_ * N_)

#define WG_ROWS   64
#define NUM_WG    (ROWS / WG_ROWS)
#define THREADS   512
#define WPITCH    272
#define APITCH    272
#define WLDS_BYTES  (G3H * WPITCH)
#define ABUF_BYTES  (WG_ROWS * APITCH)
#define TOK_BYTES   (WG_ROWS * I_ * 4)
#define SMEM_BYTES  (WLDS_BYTES + ABUF_BYTES + TOK_BYTES)

typedef __attribute__((ext_vector_type(16))) int   v16i;
typedef __attribute__((ext_vector_type(8)))  float v8f;

__device__ __forceinline__ float sigmoidf_(float x) {
  return 1.0f / (1.0f + __expf(-x));
}

__device__ __forceinline__ float tanhf_(float x) {
#if __has_builtin(__builtin_amdgcn_tanhf)
  return __builtin_amdgcn_tanhf(x);
#else
  return tanhf(x);
#endif
}

__device__ __forceinline__ unsigned f32_to_fp8(float x) {
#if __has_builtin(__builtin_amdgcn_cvt_pk_fp8_f32)
  return (unsigned)(__builtin_amdgcn_cvt_pk_fp8_f32(x, x, 0, false)) & 0xFFu;
#else
  union { float f; unsigned u; } v; v.f = x;
  unsigned s = v.u >> 31;
  float a = fabsf(x);
  if (a > 448.0f) a = 448.0f;
  if (a < 0x1.p-10f) return s << 7;
  int e; float m = frexpf(a, &e);
  int exp = e + 6;
  unsigned bits;
  if (exp <= 0) {
    int sh = 1 - exp;
    unsigned q = (unsigned)(m * 16.0f + 0.5f) >> sh;
    bits = q & 7u;
  } else {
    unsigned q = (unsigned)(m * 16.0f + 0.5f);
    if (q >= 16u) { q >>= 1; exp++; }
    unsigned mant = q - 8u;
    if (exp > 15) { exp = 15; mant = 6; }
    bits = ((unsigned)exp << 3) | (mant & 7u);
  }
  return (s << 7) | bits;
#endif
}

__global__ __launch_bounds__(THREADS, 1)
void gru_persistent(const int* __restrict__ input,
                    const float* __restrict__ W_ih,
                    const float* __restrict__ b_ih,
                    const float* __restrict__ W_hh,
                    const float* __restrict__ b_hh,
                    float* __restrict__ history)
{
  extern __shared__ char smem[];
  char* wlds = smem;
  char* abuf = smem + WLDS_BYTES;
  int*  toks = (int*)(smem + WLDS_BYTES + ABUF_BYTES);

  const int tid   = threadIdx.x;
  const int wave  = tid >> 5;
  const int lane  = tid & 31;
  const int lanelo = lane & 15;
  const int hi    = lane >> 4;
  const int row0  = blockIdx.x * WG_ROWS;

  for (int idx = tid; idx < WG_ROWS * I_; idx += THREADS)
    toks[idx] = input[row0 * I_ + idx];

  for (int idx = tid; idx < H_ * G3H; idx += THREADS) {
    int k  = idx / G3H;
    int cw = idx - k * G3H;
    ((unsigned char*)wlds)[cw * WPITCH + k] =
        (unsigned char)f32_to_fp8(W_hh[idx]);
  }

  for (int idx = tid; idx < ABUF_BYTES / 4; idx += THREADS)
    ((int*)abuf)[idx] = 0;

  const int j = wave * 16 + lanelo;
  const float bihr = b_ih[j],        bhhr = b_hh[j];
  const float bihz = b_ih[H_ + j],   bhhz = b_hh[H_ + j];
  const float bihn = b_ih[2*H_ + j], bhhn = b_hh[2*H_ + j];

  float hreg[4][8];
  #pragma unroll
  for (int mt = 0; mt < 4; ++mt)
    #pragma unroll
    for (int e = 0; e < 8; ++e) hreg[mt][e] = 0.0f;

  __syncthreads();

  for (int t = 0; t < I_; ++t) {
    v8f acc[4][3];
    #pragma unroll
    for (int mt = 0; mt < 4; ++mt)
      #pragma unroll
      for (int g = 0; g < 3; ++g)
        acc[mt][g] = (v8f){0.f,0.f,0.f,0.f,0.f,0.f,0.f,0.f};

    #pragma unroll
    for (int kh = 0; kh < 2; ++kh) {
      v16i bmat[3];
      #pragma unroll
      for (int g = 0; g < 3; ++g) {
        const char* bb = wlds
            + (size_t)(g * H_ + wave * 16 + lanelo) * WPITCH
            + kh * 128 + hi * 16;
        #pragma unroll
        for (int q = 0; q < 4; ++q) {
          int4 d = *(const int4*)(bb + q * 32);
          bmat[g][4*q+0] = d.x; bmat[g][4*q+1] = d.y;
          bmat[g][4*q+2] = d.z; bmat[g][4*q+3] = d.w;
        }
      }
      #pragma unroll
      for (int mt = 0; mt < 4; ++mt) {
        const char* ab = abuf
            + (size_t)(mt * 16 + lanelo) * APITCH
            + kh * 128 + hi * 16;
        v16i amat;
        #pragma unroll
        for (int i2 = 0; i2 < 8; ++i2) {
          uint2 d = *(const uint2*)(ab + ((i2 >> 1) * 32) + ((i2 & 1) * 8));
          amat[2*i2]   = (int)d.x;
          amat[2*i2+1] = (int)d.y;
        }
        #pragma unroll
        for (int g = 0; g < 3; ++g)
          acc[mt][g] = __builtin_amdgcn_wmma_f32_16x16x128_fp8_fp8(
              amat, bmat[g], (short)0, acc[mt][g], false, false);
      }
    }

    __syncthreads();

    #pragma unroll
    for (int mt = 0; mt < 4; ++mt) {
      #pragma unroll
      for (int e = 0; e < 8; ++e) {
        int rl  = mt * 16 + e + hi * 8;
        int tok = toks[rl * I_ + t];
        const float* wr = W_ih + (size_t)tok * G3H;
        float xr = wr[j], xz = wr[H_ + j], xn = wr[2*H_ + j];
        float r  = sigmoidf_(xr + bihr + acc[mt][0][e] + bhhr);
        float z  = sigmoidf_(xz + bihz + acc[mt][1][e] + bhhz);
        float n  = tanhf_   (xn + bihn + r * (acc[mt][2][e] + bhhn));
        float hn = (1.0f - z) * n + z * hreg[mt][e];
        hreg[mt][e] = hn;
        history[((size_t)(row0 + rl) * I_ + t) * H_ + j] = hn;
        ((unsigned char*)abuf)[rl * APITCH + j] = (unsigned char)f32_to_fp8(hn);
      }
    }
    __syncthreads();
  }
}

__global__ __launch_bounds__(H_)
void gru_finalize(const float* __restrict__ history,
                  const int* __restrict__ lengths,
                  float* __restrict__ last)
{
  int b = blockIdx.x;
  int h = threadIdx.x;
  float s = 0.0f;
  #pragma unroll
  for (int n = 0; n < N_; ++n) {
    int row = b * N_ + n;
    int t   = lengths[row] - 1;
    s += history[((size_t)row * I_ + t) * H_ + h];
  }
  last[b * H_ + h] = s * (1.0f / (float)N_);
}

extern "C" void kernel_launch(void* const* d_in, const int* in_sizes, int n_in,
                              void* d_out, int out_size, void* d_ws, size_t ws_size,
                              hipStream_t stream) {
  (void)in_sizes; (void)n_in; (void)out_size; (void)d_ws; (void)ws_size;
  const int*   input   = (const int*)  d_in[0];
  const int*   lengths = (const int*)  d_in[1];
  const float* W_ih    = (const float*)d_in[2];
  const float* b_ih    = (const float*)d_in[3];
  const float* W_hh    = (const float*)d_in[4];
  const float* b_hh    = (const float*)d_in[5];

  float* history = (float*)d_out;
  float* last    = history + (size_t)B_ * N_ * I_ * H_;

  (void)hipFuncSetAttribute((const void*)gru_persistent,
                            hipFuncAttributeMaxDynamicSharedMemorySize,
                            (int)SMEM_BYTES);

  gru_persistent<<<dim3(NUM_WG), dim3(THREADS), SMEM_BYTES, stream>>>(
      input, W_ih, b_ih, W_hh, b_hh, history);
  gru_finalize<<<dim3(B_), dim3(H_), 0, stream>>>(history, lengths, last);
}
